// EMG_15101105013515
// MI455X (gfx1250) — matrix operand order repaired and checked by reading, not compiled, not run on hardware
//
#include <hip/hip_runtime.h>

#define B_   32
#define T_   1024
#define H_   512
#define L_   2
#define TWOH (2 * H_)
#define NBLK 16
#define TPB  128

static constexpr size_t OUTSZ = (size_t)B_ * T_ * L_ * H_;

typedef _Float16 h16;
typedef h16   v16h __attribute__((ext_vector_type(16)));
typedef h16   v8h  __attribute__((ext_vector_type(8)));
typedef float v8f  __attribute__((ext_vector_type(8)));

// Raw v_rcp_f32 (1 ulp-ish) -- good enough for gate nonlinearities and far
// cheaper than the IEEE div_scale/div_fmas sequence the compiler emits for '/'.
__device__ __forceinline__ float fast_rcp(float x) {
  return __builtin_amdgcn_rcpf(x);
}

__device__ __forceinline__ float sigmoidf_(float v) {
  // 1 / (1 + e^-v):  v_exp_f32 + v_add + v_rcp
  return fast_rcp(1.0f + __expf(-v));
}

__device__ __forceinline__ float tanhf_(float x) {
  // tanh(x) = 1 - 2/(1 + e^{2x}); exp overflow -> rcp(inf)=0 -> +1,
  // exp underflow -> 1 - 2 = -1. Correct saturation at both ends.
  float e = __expf(2.0f * x);
  return 1.0f - 2.0f * fast_rcp(1.0f + e);
}

// Sense-reversal grid barrier: bar[0]=arrive counter, bar[1]=generation.
__device__ __forceinline__ void grid_barrier(unsigned* bar) {
  __syncthreads();
  if (threadIdx.x == 0) {
    __threadfence();  // make this block's global writes visible device-wide
    unsigned* cnt = bar;
    unsigned* gen = bar + 1;
    unsigned g = __hip_atomic_load(gen, __ATOMIC_RELAXED, __HIP_MEMORY_SCOPE_AGENT);
    unsigned a = __hip_atomic_fetch_add(cnt, 1u, __ATOMIC_ACQ_REL, __HIP_MEMORY_SCOPE_AGENT);
    if (a == NBLK - 1) {
      __hip_atomic_store(cnt, 0u, __ATOMIC_RELAXED, __HIP_MEMORY_SCOPE_AGENT);
      __hip_atomic_store(gen, g + 1u, __ATOMIC_RELEASE, __HIP_MEMORY_SCOPE_AGENT);
    } else {
      while (__hip_atomic_load(gen, __ATOMIC_ACQUIRE, __HIP_MEMORY_SCOPE_AGENT) == g) {
        __builtin_amdgcn_s_sleep(2);
      }
    }
  }
  __syncthreads();
}

// One wave computes two 16x16 gate tiles (rg cols j..j+15, mg cols H+j..H+j+15)
// for one 16-row M-tile, accumulating over K=2H with v_wmma_f32_16x16x32_f16.
__device__ __forceinline__ void gemm_tile(const h16* __restrict__ aP,
                                          const h16* __restrict__ wR,
                                          const h16* __restrict__ wM,
                                          int koffA, v8f& accR, v8f& accM) {
#pragma unroll 4
  for (int k = 0; k < TWOH; k += 32) {
    // A fragment (16-bit 16x32): lo lanes K 0..15, hi lanes K 16..31 (contiguous)
    v8h alo = *(const v8h*)(aP + k + koffA);
    v8h ahi = *(const v8h*)(aP + k + 8 + koffA);
    v16h a = __builtin_shufflevector(alo, ahi, 0, 1, 2, 3, 4, 5, 6, 7,
                                     8, 9, 10, 11, 12, 13, 14, 15);
    // B fragment (32x16, column = lane&15 = row of W, contiguous in K)
    v16h bR = *(const v16h*)(wR + k);
    v16h bM = *(const v16h*)(wM + k);
    accR = __builtin_amdgcn_wmma_f32_16x16x32_f16(false, a, false, bR,
                                                  (short)0, accR, false, false);
    accM = __builtin_amdgcn_wmma_f32_16x16x32_f16(false, a, false, bM,
                                                  (short)0, accM, false, false);
  }
}

// Convert W (f32 -> f16, stays hot in L2 for all 1024 steps) and reset barrier.
__global__ void emg_init(const float* __restrict__ W, h16* __restrict__ Whf,
                         unsigned* __restrict__ bar, int n) {
  int i = blockIdx.x * blockDim.x + threadIdx.x;
  if (i == 0) { bar[0] = 0u; bar[1] = 0u; }
  int stride = gridDim.x * blockDim.x;
  for (; i < n; i += stride) Whf[i] = (h16)W[i];
}

__global__ __launch_bounds__(TPB, 1) void emg_persistent(
    const float* __restrict__ x, const float* __restrict__ bias,
    const h16* __restrict__ Whf, float* __restrict__ out,
    h16* __restrict__ mg0a, h16* __restrict__ mg0b,
    h16* __restrict__ mg1a, h16* __restrict__ mg1b,
    unsigned* __restrict__ bar) {
  const int lane  = threadIdx.x & 31;
  const int wave  = threadIdx.x >> 5;
  const int unit  = blockIdx.x * (TPB / 32) + wave;  // 0..63
  const int mtile = unit >> 5;                       // 0..1  (rows 0-15 / 16-31)
  const int ntile = unit & 31;                       // 0..31 (cols 16n..16n+15)
  const int jn    = lane & 15;
  const bool hi   = lane >= 16;
  const int j     = ntile * 16 + jn;

  // ---- init t=0 merge buffers: merge0 = [x_0 (f16), m0=0]; merge1 m-half = 0
  {
    int gtid = blockIdx.x * TPB + threadIdx.x;
    for (int i = gtid; i < B_ * H_; i += NBLK * TPB) {
      int b = i >> 9;        // / H_
      int h = i & (H_ - 1);  // % H_
      mg0a[b * TWOH + h]      = (h16)x[((size_t)b * T_) * H_ + h];
      mg0a[b * TWOH + H_ + h] = (h16)0.0f;
      mg1a[b * TWOH + h]      = (h16)0.0f;
      mg1a[b * TWOH + H_ + h] = (h16)0.0f;
    }
  }
  grid_barrier(bar);

  const float b0R = bias[j];
  const float b0M = bias[H_ + j];
  const float b1R = bias[TWOH + j];
  const float b1M = bias[TWOH + H_ + j];

  const int koffB = hi ? 16 : 0;
  const h16* w0R = Whf + (size_t)j * TWOH + koffB;
  const h16* w0M = Whf + (size_t)(H_ + j) * TWOH + koffB;
  const h16* w1R = Whf + (size_t)TWOH * TWOH + (size_t)j * TWOH + koffB;
  const h16* w1M = Whf + (size_t)TWOH * TWOH + (size_t)(H_ + j) * TWOH + koffB;

  const int arow  = mtile * 16 + jn;  // A-frag row (lanes 0-15 / 16-31 same M)
  const int koffA = hi ? 16 : 0;

  // Recurrent cell state + layer-0 output live in registers for all T steps.
  float c0[8], c1[8], inp1[8];
#pragma unroll
  for (int r = 0; r < 8; ++r) { c0[r] = 0.0f; c1[r] = 0.0f; }

  const v8f vzero = {0.f, 0.f, 0.f, 0.f, 0.f, 0.f, 0.f, 0.f};

  for (int t = 0; t < T_; ++t) {
    h16* m0cur = (t & 1) ? mg0b : mg0a;
    h16* m0nxt = (t & 1) ? mg0a : mg0b;
    h16* m1cur = (t & 1) ? mg1b : mg1a;
    h16* m1nxt = (t & 1) ? mg1a : mg1b;

    // ================= layer 0 =================
    v8f accR = vzero, accM = vzero;
    gemm_tile(m0cur + (size_t)arow * TWOH, w0R, w0M, koffA, accR, accM);

#pragma unroll
    for (int r = 0; r < 8; ++r) {
      int row = mtile * 16 + (hi ? r + 8 : r);  // C/D layout: M=r / r+8
      float rg  = sigmoidf_(accR[r] + b0R);
      float mgt = sigmoidf_(accM[r] + b0M);
      float inp = x[((size_t)row * T_ + t) * H_ + j];
      float cn  = tanhf_(c0[r] + rg * inp);
      float mn  = mgt * inp + (1.0f - mgt) * cn;
      c0[r]   = cn;
      inp1[r] = mn;  // layer-1 input, same lane — stays in registers
      size_t oidx = (((size_t)row * T_ + t) * L_ + 0) * H_ + j;
      out[oidx]         = mn;
      out[OUTSZ + oidx] = mn;  // duplicated output (tuple returns output twice)
      h16 mh = (h16)mn;
      m1cur[row * TWOH + j]      = mh;  // input half for layer-1 GEMM this step
      m0nxt[row * TWOH + H_ + j] = mh;  // m0-state half for next step
    }
    grid_barrier(bar);

    // ================= layer 1 =================
    accR = vzero; accM = vzero;
    gemm_tile(m1cur + (size_t)arow * TWOH, w1R, w1M, koffA, accR, accM);

#pragma unroll
    for (int r = 0; r < 8; ++r) {
      int row = mtile * 16 + (hi ? r + 8 : r);
      float rg  = sigmoidf_(accR[r] + b1R);
      float mgt = sigmoidf_(accM[r] + b1M);
      float inp = inp1[r];
      float cn  = tanhf_(c1[r] + rg * inp);
      float mn  = mgt * inp + (1.0f - mgt) * cn;
      c1[r] = cn;
      size_t oidx = (((size_t)row * T_ + t) * L_ + 1) * H_ + j;
      out[oidx]         = mn;
      out[OUTSZ + oidx] = mn;
      m1nxt[row * TWOH + H_ + j] = (h16)mn;  // m1-state half for next step
      if (t + 1 < T_)                        // prefill x_{t+1} into next merge0
        m0nxt[row * TWOH + j] = (h16)x[((size_t)row * T_ + (t + 1)) * H_ + j];
    }
    grid_barrier(bar);
  }

  // ============ con: final cell states broadcast over T (from registers) =====
  for (int t = 0; t < T_; ++t) {
#pragma unroll
    for (int r = 0; r < 8; ++r) {
      int row = mtile * 16 + (hi ? r + 8 : r);
      size_t base = 2 * OUTSZ + (((size_t)row * T_ + t) * L_) * H_ + j;
      out[base]      = c0[r];
      out[base + H_] = c1[r];
    }
  }
}

extern "C" void kernel_launch(void* const* d_in, const int* in_sizes, int n_in,
                              void* d_out, int out_size, void* d_ws, size_t ws_size,
                              hipStream_t stream) {
  const float* x    = (const float*)d_in[0];
  const float* W    = (const float*)d_in[1];
  const float* bias = (const float*)d_in[2];
  float* out = (float*)d_out;

  // ws layout: [0,256): barrier; [256, 256+4MB): W in f16; then 4 merge bufs
  char* ws = (char*)d_ws;
  unsigned* bar = (unsigned*)ws;
  h16* Whf = (h16*)(ws + 256);
  const size_t welems = (size_t)L_ * TWOH * TWOH;  // 2M elements = 4 MB f16
  h16* merge = (h16*)(ws + 256 + welems * sizeof(h16));
  const size_t msz = (size_t)B_ * TWOH;  // 32768 elems = 64 KB each
  h16* mg0a = merge;
  h16* mg0b = merge + msz;
  h16* mg1a = merge + 2 * msz;
  h16* mg1b = merge + 3 * msz;

  emg_init<<<1024, 256, 0, stream>>>(W, Whf, bar, (int)welems);
  emg_persistent<<<NBLK, TPB, 0, stream>>>(x, bias, Whf, out,
                                           mg0a, mg0b, mg1a, mg1b, bar);
}
